// SelfAttention_73349451481684
// MI455X (gfx1250) — compile-verified, matrix operand order checked by reading, not run on hardware
//
#include <hip/hip_runtime.h>
#include <hip/hip_bf16.h>
#include <math.h>

// ---------------------------------------------------------------------------
// Types for CDNA5 WMMA (wave32)
// ---------------------------------------------------------------------------
typedef __bf16 bf16_t;
typedef __attribute__((ext_vector_type(16))) __bf16 v16bf;
typedef __attribute__((ext_vector_type(8)))  __bf16 v8bf;
typedef __attribute__((ext_vector_type(4)))  __bf16 v4bf;
typedef __attribute__((ext_vector_type(8)))  float  v8f;

#define HIDDEN   2048
#define KV_DIM   512
#define SEQ      2048
#define HEAD_DIM 64
#define NHEADS   32

// ---------------------------------------------------------------------------
// Fragment loader: 16x32 bf16 tile, row-major, rowStride multiple of 8.
// CDNA5 16-bit A-operand layout (B operand = same layout of N-major transpose):
//   lanes 0-15 : M = lane,    halfs 0-7 -> K 0..7,  halfs 8-15 -> K 16..23
//   lanes 16-31: M = lane-16, halfs 0-7 -> K 8..15, halfs 8-15 -> K 24..31
// ---------------------------------------------------------------------------
__device__ inline v16bf load_frag16x32(const bf16_t* __restrict__ base,
                                       int rowStride, int lane) {
  const int r  = lane & 15;
  const int k0 = (lane < 16) ? 0 : 8;
  const bf16_t* p = base + r * rowStride;
  v8bf lo = *reinterpret_cast<const v8bf*>(p + k0);
  v8bf hi = *reinterpret_cast<const v8bf*>(p + k0 + 16);
  return __builtin_shufflevector(lo, hi, 0, 1, 2, 3, 4, 5, 6, 7,
                                         8, 9, 10, 11, 12, 13, 14, 15);
}

__device__ inline v8f wmma_bf16(v16bf a, v16bf b, v8f c) {
  return __builtin_amdgcn_wmma_f32_16x16x32_bf16(
      false, a, false, b, (short)0, c, false, false);
}

// ---------------------------------------------------------------------------
// Kernel 1: fused QKV projection with ping-pong double-buffered LDS and a
// one-tile-ahead register pipeline (single barrier per K-step).
//   C[2048, 3072] = x @ [Wq | Wk | Wv] + [bq | bk | bv]
// Q (bf16 row-major), K (bf16 row-major), V stored transposed Vt[512,2048].
// 128x128 tile/WG, 8 waves as 2(M) x 4(N), K-step 32.
// ---------------------------------------------------------------------------
#define T1_LDS 40   // LDS row stride in halfs (multiple of 8, padded vs 32)

__global__ __launch_bounds__(256)
void qkv_gemm_kernel(const float* __restrict__ x,
                     const float* __restrict__ Wq, const float* __restrict__ bq,
                     const float* __restrict__ Wk, const float* __restrict__ bk,
                     const float* __restrict__ Wv, const float* __restrict__ bv,
                     bf16_t* __restrict__ Qo, bf16_t* __restrict__ Ko,
                     bf16_t* __restrict__ Vt) {
  __shared__ bf16_t ldsA[2][128 * T1_LDS];   // x tile   : 128 rows x 32 k
  __shared__ bf16_t ldsB[2][128 * T1_LDS];   // W tile ^T: 128 cols x 32 k

  const int n0 = blockIdx.x * 128;        // output column base (0..3071)
  const int m0 = blockIdx.y * 128;        // output row base

  // Uniform (SGPR) selection of weight matrix and unified output addressing:
  //   out element (row, col) -> obase[row*ors + col*ocs]   (32-bit offsets)
  const float* W;  const float* bias;  int ldW; int nloc;
  bf16_t* obase;   int ors, ocs;
  if (n0 < HIDDEN) {
    W = Wq; bias = bq; ldW = HIDDEN; nloc = n0;
    obase = Qo; ors = HIDDEN; ocs = 1;
  } else if (n0 < HIDDEN + 512) {
    W = Wk; bias = bk; ldW = KV_DIM; nloc = n0 - HIDDEN;
    obase = Ko; ors = KV_DIM; ocs = 1;
  } else {
    W = Wv; bias = bv; ldW = KV_DIM; nloc = n0 - HIDDEN - 512;
    obase = Vt; ors = 1; ocs = SEQ;            // transposed V
  }

  const int tid   = threadIdx.x;
  const int lane  = tid & 31;
  const int wave  = tid >> 5;
  const int waveM = wave >> 2;            // 0..1 -> 64 rows each
  const int waveN = wave & 3;             // 0..3 -> 32 cols each

  // One-tile-ahead register pipeline.
  float4 aReg[4], bReg[4];

  auto fetch = [&](int kk) {
#pragma unroll
    for (int j = 0; j < 4; ++j) {
      int la = tid + j * 256;
      int arow = la >> 3, aq = la & 7;
      aReg[j] = *reinterpret_cast<const float4*>(
          x + (size_t)((m0 + arow) * HIDDEN + kk + 4 * aq));
      int krow = la >> 5, bq_ = la & 31;
      bReg[j] = *reinterpret_cast<const float4*>(
          W + (size_t)((kk + krow) * ldW + nloc + 4 * bq_));
    }
  };
  auto commit = [&](int buf) {
#pragma unroll
    for (int j = 0; j < 4; ++j) {
      int la = tid + j * 256;
      int arow = la >> 3, aq = la & 7;
      v4bf pk = {(bf16_t)aReg[j].x, (bf16_t)aReg[j].y,
                 (bf16_t)aReg[j].z, (bf16_t)aReg[j].w};
      *reinterpret_cast<v4bf*>(&ldsA[buf][arow * T1_LDS + 4 * aq]) = pk;
      int krow = la >> 5, bq_ = la & 31;
      ldsB[buf][(4 * bq_ + 0) * T1_LDS + krow] = (bf16_t)bReg[j].x;
      ldsB[buf][(4 * bq_ + 1) * T1_LDS + krow] = (bf16_t)bReg[j].y;
      ldsB[buf][(4 * bq_ + 2) * T1_LDS + krow] = (bf16_t)bReg[j].z;
      ldsB[buf][(4 * bq_ + 3) * T1_LDS + krow] = (bf16_t)bReg[j].w;
    }
  };

  v8f acc[4][2];
#pragma unroll
  for (int i = 0; i < 4; ++i)
#pragma unroll
    for (int j = 0; j < 2; ++j) acc[i][j] = {};

  const int NSTEP = HIDDEN / 32;          // 64
  fetch(0);
  commit(0);
  fetch(32);                              // tile 1 in flight during compute 0
  __syncthreads();                        // buf0 published

  int cur = 0;
  for (int i = 0; i < NSTEP; ++i) {
    // ---- compute tile i from buf[cur] (overlaps outstanding fetch) ----
    v16bf bfrag[2];
#pragma unroll
    for (int nj = 0; nj < 2; ++nj)
      bfrag[nj] = load_frag16x32(&ldsB[cur][(waveN * 32 + nj * 16) * T1_LDS],
                                 T1_LDS, lane);
#pragma unroll
    for (int mi = 0; mi < 4; ++mi) {
      v16bf afrag = load_frag16x32(&ldsA[cur][(waveM * 64 + mi * 16) * T1_LDS],
                                   T1_LDS, lane);
#pragma unroll
      for (int nj = 0; nj < 2; ++nj)
        acc[mi][nj] = wmma_bf16(afrag, bfrag[nj], acc[mi][nj]);
    }
    // ---- publish tile i+1 into the other buffer, start fetch of i+2 ----
    if (i + 1 < NSTEP) {
      commit(cur ^ 1);                    // waits on fetch(i+1) loads
      if (i + 2 < NSTEP) fetch((i + 2) * 32);
    }
    __syncthreads();
    cur ^= 1;
  }

  // ---- epilogue: branch-free unified store, 32-bit offset math ----
  const int rowOff = (lane >= 16) ? 8 : 0;
  const int nl     = lane & 15;
#pragma unroll
  for (int mi = 0; mi < 4; ++mi) {
#pragma unroll
    for (int nj = 0; nj < 2; ++nj) {
      int colLocal = nloc + waveN * 32 + nj * 16 + nl;
      float bv_ = bias[colLocal];
      int cterm = colLocal * ocs;
      int rbase = (m0 + waveM * 64 + mi * 16 + rowOff) * ors + cterm;
#pragma unroll
      for (int r = 0; r < 8; ++r)
        obase[(size_t)(rbase + r * ors)] = (bf16_t)(acc[mi][nj][r] + bv_);
    }
  }
}

// ---------------------------------------------------------------------------
// Kernel 2: GQA flash attention + residual.
// grid = (32 heads, 8 q-blocks of 256 rows); 8 waves/WG, 32 q-rows per wave
// (two 16-row M-tiles share each K/V fragment -> 16 wmma per 32-key chunk).
// Q/K fragments load straight from global; Vt (pre-transposed) likewise.
// Only P (probabilities) bounces through wave-private LDS (C->A layout).
// Softmax runs in the log2 domain: scores pre-scaled by 0.125*log2(e).
// ---------------------------------------------------------------------------
#define P_LDS 40

__global__ __launch_bounds__(256)
void attention_kernel(const float* __restrict__ x,
                      const bf16_t* __restrict__ Q,
                      const bf16_t* __restrict__ K,
                      const bf16_t* __restrict__ Vt,
                      float* __restrict__ out) {
  __shared__ bf16_t ldsP[8 * 2 * 16 * P_LDS];

  const int h    = blockIdx.x;            // head 0..31
  const int qb   = blockIdx.y;            // q block 0..7 (256 rows each)
  const int wave = threadIdx.x >> 5;
  const int lane = threadIdx.x & 31;
  const int g    = h >> 2;                // KV group (rep = 4)
  const int qRow = qb * 256 + wave * 32;  // this wave's 32 query rows

  // 1/sqrt(64) * log2(e): scores enter the log2 domain directly.
  const float C = 0.125f * 1.44269504088896f;

  // Q fragments: [mi][dim-piece]
  v16bf qf[2][2];
#pragma unroll
  for (int mi = 0; mi < 2; ++mi) {
    const bf16_t* Qb = Q + (size_t)((qRow + mi * 16) * HIDDEN + h * HEAD_DIM);
    qf[mi][0] = load_frag16x32(Qb,      HIDDEN, lane);
    qf[mi][1] = load_frag16x32(Qb + 32, HIDDEN, lane);
  }

  v8f o[2][4];
  float mrow[2][8], lrow[2][8];
#pragma unroll
  for (int mi = 0; mi < 2; ++mi) {
#pragma unroll
    for (int t = 0; t < 4; ++t) o[mi][t] = {};
#pragma unroll
    for (int r = 0; r < 8; ++r) { mrow[mi][r] = -INFINITY; lrow[mi][r] = 0.0f; }
  }

  bf16_t* Pw = &ldsP[wave * 2 * 16 * P_LDS];
  const int rowOff = (lane >= 16) ? 8 : 0;
  const int nl     = lane & 15;

  const bf16_t* Khead = K + g * HEAD_DIM;
  const bf16_t* Vhead = Vt + g * HEAD_DIM * SEQ;

  for (int c = 0; c < SEQ; c += 32) {
    // prefetch next chunk's K rows / Vt rows into cache (speculative)
    {
      int cn = (c + 32 < SEQ) ? (c + 32) : c;
      __builtin_prefetch(Khead + (cn + lane) * KV_DIM, 0, 3);
      __builtin_prefetch(Vhead + (lane & 15) * SEQ + cn + ((lane >> 4) << 4),
                         0, 3);
    }

    // ---- K fragments for this 32-key chunk: [key-tile][dim-piece] ----
    const bf16_t* Kb = Khead + c * KV_DIM;
    v16bf kf[2][2];
    kf[0][0] = load_frag16x32(Kb,                    KV_DIM, lane);
    kf[0][1] = load_frag16x32(Kb + 32,               KV_DIM, lane);
    kf[1][0] = load_frag16x32(Kb + 16 * KV_DIM,      KV_DIM, lane);
    kf[1][1] = load_frag16x32(Kb + 16 * KV_DIM + 32, KV_DIM, lane);

    // ---- scores for both M-tiles ----
    v8f s[2][2];
#pragma unroll
    for (int mi = 0; mi < 2; ++mi) {
#pragma unroll
      for (int kt = 0; kt < 2; ++kt) {
        v8f z = {};
        z = wmma_bf16(qf[mi][0], kf[kt][0], z);
        z = wmma_bf16(qf[mi][1], kf[kt][1], z);
        s[mi][kt] = z;
      }
    }

    // ---- online softmax (log2 domain) per M-tile; write P to LDS ----
#pragma unroll
    for (int mi = 0; mi < 2; ++mi) {
      bf16_t* P = Pw + mi * 16 * P_LDS;
      float alpha[8];
#pragma unroll
      for (int r = 0; r < 8; ++r) {
        float a = s[mi][0][r] * C;
        float b = s[mi][1][r] * C;
        float t = fmaxf(a, b);
        t = fmaxf(t, __shfl_xor(t, 1));
        t = fmaxf(t, __shfl_xor(t, 2));
        t = fmaxf(t, __shfl_xor(t, 4));
        t = fmaxf(t, __shfl_xor(t, 8));
        float mn = fmaxf(mrow[mi][r], t);
        alpha[r] = exp2f(mrow[mi][r] - mn);
        float p0 = exp2f(a - mn);
        float p1 = exp2f(b - mn);
        float rs = p0 + p1;
        rs += __shfl_xor(rs, 1);
        rs += __shfl_xor(rs, 2);
        rs += __shfl_xor(rs, 4);
        rs += __shfl_xor(rs, 8);
        lrow[mi][r] = lrow[mi][r] * alpha[r] + rs;
        mrow[mi][r] = mn;
        P[(r + rowOff) * P_LDS + nl]      = (bf16_t)p0;
        P[(r + rowOff) * P_LDS + 16 + nl] = (bf16_t)p1;
      }
#pragma unroll
      for (int r = 0; r < 8; ++r) {
        o[mi][0][r] *= alpha[r]; o[mi][1][r] *= alpha[r];
        o[mi][2][r] *= alpha[r]; o[mi][3][r] *= alpha[r];
      }
    }
#if __has_builtin(__builtin_amdgcn_wave_barrier)
    __builtin_amdgcn_wave_barrier();
#endif

    // ---- V fragments (shared by both M-tiles), then P.V ----
    const bf16_t* Vb = Vhead + c;
    v16bf vf[4];
#pragma unroll
    for (int t = 0; t < 4; ++t)
      vf[t] = load_frag16x32(Vb + t * 16 * SEQ, SEQ, lane);

#pragma unroll
    for (int mi = 0; mi < 2; ++mi) {
      v16bf pf = load_frag16x32(Pw + mi * 16 * P_LDS, P_LDS, lane);
#pragma unroll
      for (int t = 0; t < 4; ++t)
        o[mi][t] = wmma_bf16(pf, vf[t], o[mi][t]);
    }
  }

  // ---- normalize, residual add, store f32 (32-bit offset math) ----
#pragma unroll
  for (int mi = 0; mi < 2; ++mi) {
#pragma unroll
    for (int r = 0; r < 8; ++r) {
      float inv = 1.0f / lrow[mi][r];
      int row = qRow + mi * 16 + r + rowOff;
      int base = row * HIDDEN + h * HEAD_DIM + nl;
      out[(size_t)(base + 0)]  = o[mi][0][r] * inv + x[(size_t)(base + 0)];
      out[(size_t)(base + 16)] = o[mi][1][r] * inv + x[(size_t)(base + 16)];
      out[(size_t)(base + 32)] = o[mi][2][r] * inv + x[(size_t)(base + 32)];
      out[(size_t)(base + 48)] = o[mi][3][r] * inv + x[(size_t)(base + 48)];
    }
  }
}

// ---------------------------------------------------------------------------
// Launch
// ---------------------------------------------------------------------------
extern "C" void kernel_launch(void* const* d_in, const int* in_sizes, int n_in,
                              void* d_out, int out_size, void* d_ws, size_t ws_size,
                              hipStream_t stream) {
  const float* x  = (const float*)d_in[0];
  const float* Wq = (const float*)d_in[1];
  const float* bq = (const float*)d_in[2];
  const float* Wk = (const float*)d_in[3];
  const float* bk = (const float*)d_in[4];
  const float* Wv = (const float*)d_in[5];
  const float* bv = (const float*)d_in[6];
  float* out = (float*)d_out;

  // workspace: Q[2048,2048] | K[2048,512] | Vt[512,2048], all bf16 = 12 MB
  bf16_t* Qw  = (bf16_t*)d_ws;
  bf16_t* Kw  = Qw + (size_t)SEQ * HIDDEN;
  bf16_t* Vtw = Kw + (size_t)SEQ * KV_DIM;

  dim3 g1((HIDDEN + 2 * KV_DIM) / 128, SEQ / 128);   // (24, 16)
  qkv_gemm_kernel<<<g1, 256, 0, stream>>>(x, Wq, bq, Wk, bk, Wv, bv,
                                          Qw, Kw, Vtw);

  dim3 g2(NHEADS, SEQ / 256);                        // (32, 8)
  attention_kernel<<<g2, 256, 0, stream>>>(x, Qw, Kw, Vtw, out);
}
